// TopoDPConv_82497731821730
// MI455X (gfx1250) — matrix operand order repaired and checked by reading, not compiled, not run on hardware
//
#include <hip/hip_runtime.h>
#include <hip/hip_bf16.h>

typedef __attribute__((ext_vector_type(16))) _Float16 v16h;
typedef __attribute__((ext_vector_type(8)))  float    v8f;

#define N_NODES 8192
#define DEG     16
#define HID     128
#define IN_DIM  128
#define ED      8
#define KZ      160   // 136 padded to 5*32

// ---------------- WMMA fragment helpers (ISA 7.12.2 layouts) ----------------
// A: 16x32 (MxK) f16. lane l<16: row l, halves = K {k0+0..7, k0+16..23};
//                     lane l>=16: row l-16, halves = K {k0+8..15, k0+24..31}.
__device__ __forceinline__ v16h load_a_frag(const _Float16* base, int stride,
                                            int k0, int lane) {
  const int row = lane & 15;
  const int kb  = k0 + ((lane & 16) ? 8 : 0);
  const _Float16* p = base + row * stride + kb;
  v16h a;
#pragma unroll
  for (int i = 0; i < 8; ++i) { a[i] = p[i]; a[8 + i] = p[16 + i]; }
  return a;
}

// A fragment for an M=1 GEMV: row 0 = vec[k], rows 1..15 = 0.
// Only lanes 0 and 16 carry data (both map to row 0).
__device__ __forceinline__ v16h load_a_frag_vec(const _Float16* vec, int k0,
                                                int lane) {
  v16h a;
  if ((lane & 15) == 0) {
    const int kb = k0 + ((lane & 16) ? 8 : 0);
#pragma unroll
    for (int i = 0; i < 8; ++i) { a[i] = vec[kb + i]; a[8 + i] = vec[kb + 16 + i]; }
  } else {
#pragma unroll
    for (int i = 0; i < 16; ++i) a[i] = (_Float16)0.0f;
  }
  return a;
}

// B: 32x16 (KxN) f16, stored as W^T (n-major, contiguous K).
// lane l<16: col n0+l, K = {k0+0..7, k0+16..23};  lane l>=16: col n0+l-16, K = {k0+8..15, k0+24..31}.
__device__ __forceinline__ v16h load_b_frag(const _Float16* wT, int strideK,
                                            int n0, int k0, int lane) {
  const int col = n0 + (lane & 15);
  const int kb  = k0 + ((lane & 16) ? 8 : 0);
  const _Float16* p = wT + col * strideK + kb;
  v16h b;
#pragma unroll
  for (int i = 0; i < 8; ++i) { b[i] = p[i]; b[8 + i] = p[16 + i]; }
  return b;
}

// C/D 16x16 f32: lane l<16 holds (M=r, N=l) in vgpr r; lane l>=16 holds (M=r+8, N=l-16).

// ---------------- prep: f16-transposed weights into workspace ----------------
// ws layout (halves):
//   WpT  [128*128] @ 0      Wm1T [128*160] @ 16384   Wm2T [128*128] @ 36864
//   Wu1T [128*256] @ 53248  Wu2T [128*128] @ 86016   (total 102400 halves)
__global__ void prep_kernel(const float* __restrict__ Wp,
                            const float* __restrict__ Wm1,
                            const float* __restrict__ Wm2,
                            const float* __restrict__ Wu1,
                            const float* __restrict__ Wu2,
                            _Float16* __restrict__ wsH) {
  int i = blockIdx.x * blockDim.x + threadIdx.x;            // 400*256 = 102400
  if (i < 16384) {                                          // WpT[n][k]
    int n = i >> 7, k = i & 127;
    wsH[n * 128 + k] = (_Float16)Wp[k * 128 + n];
  } else if (i < 36864) {                                   // Wm1T[n][kk], K padded
    int j = i - 16384;
    int n = j / KZ, kk = j % KZ;
    float v = (kk < HID + ED) ? Wm1[kk * 128 + n] : 0.0f;
    wsH[16384 + n * KZ + kk] = (_Float16)v;
  } else if (i < 53248) {                                   // Wm2T[n][k]
    int j = i - 36864;
    int n = j >> 7, k = j & 127;
    wsH[36864 + n * 128 + k] = (_Float16)Wm2[k * 128 + n];
  } else if (i < 86016) {                                   // Wu1T[n][k], K=256
    int j = i - 53248;
    int n = j >> 8, k = j & 255;
    wsH[53248 + n * 256 + k] = (_Float16)Wu1[k * 128 + n];
  } else if (i < 102400) {                                  // Wu2T[n][k]
    int j = i - 86016;
    int n = j >> 7, k = j & 127;
    wsH[86016 + n * 128 + k] = (_Float16)Wu2[k * 128 + n];
  }
}

// ---------------- h0 = x @ Wp + bp  (WMMA tiled GEMM) ----------------
__global__ void h0_kernel(const float* __restrict__ x,
                          const _Float16* __restrict__ WpT,
                          const float* __restrict__ bp,
                          float* __restrict__ out) {
  __shared__ _Float16 sX[16 * 128];
  const int tid = threadIdx.x;
  const int r0  = blockIdx.x * 16;
  for (int i = tid; i < 16 * 128; i += 256) {
    int r = i >> 7, c = i & 127;
    sX[i] = (_Float16)x[(r0 + r) * 128 + c];
  }
  __syncthreads();

  const int w = tid >> 5, lane = tid & 31;
  v8f acc = {};
#pragma unroll
  for (int kt = 0; kt < 4; ++kt) {
    v16h a = load_a_frag(sX, 128, kt * 32, lane);
    v16h b = load_b_frag(WpT, 128, w * 16, kt * 32, lane);
    acc = __builtin_amdgcn_wmma_f32_16x16x32_f16(false, a, false, b,
                                                 (short)0, acc, false, false);
  }
  const int n = w * 16 + (lane & 15);
  const float bias = bp[n];
  const int rbase = (lane & 16) ? 8 : 0;
#pragma unroll
  for (int r = 0; r < 8; ++r)
    out[(r0 + rbase + r) * 128 + n] = acc[r] + bias;
}

// ---------------- sequential topological scan (single workgroup) ----------------
__global__ __launch_bounds__(256, 1)
void scan_kernel(float* __restrict__ h,                 // d_out, init = h0
                 const int* __restrict__ ei,            // edge_index [2,E], src row first
                 const float* __restrict__ attr,        // [E, ED]
                 const float* __restrict__ bm1, const float* __restrict__ bm2,
                 const float* __restrict__ bu1, const float* __restrict__ bu2,
                 const _Float16* __restrict__ wsH) {
  // all weights live in LDS (~182 KB of the WGP's 320 KB)
  __shared__ _Float16 sWm1T[128 * KZ];    // 40960 B
  __shared__ _Float16 sWm2T[128 * 128];   // 32768 B
  __shared__ _Float16 sWu1T[128 * 256];   // 65536 B
  __shared__ _Float16 sWu2T[128 * 128];   // 32768 B
  __shared__ _Float16 sZ[16 * KZ];        //  5120 B
  __shared__ _Float16 sMsg1[16 * 128];    //  4096 B
  __shared__ _Float16 sCat[256];          //   512 B   [h[v] | agg] as f16
  __shared__ _Float16 sU1[128];           //   256 B

  const int tid  = threadIdx.x;
  const int w    = tid >> 5;
  const int lane = tid & 31;

  // stage all MLP weights into LDS once
  for (int i = tid; i < 128 * KZ; i += 256)  sWm1T[i] = wsH[16384 + i];
  for (int i = tid; i < 128 * 128; i += 256) sWm2T[i] = wsH[36864 + i];
  for (int i = tid; i < 128 * 256; i += 256) sWu1T[i] = wsH[53248 + i];
  for (int i = tid; i < 128 * 128; i += 256) sWu2T[i] = wsH[86016 + i];

  // per-thread loop invariants
  const int   n     = w * 16 + (lane & 15);
  const int   rbase = (lane & 16) ? 8 : 0;
  const float bias1  = bm1[n];
  const float bias2  = bm2[n];
  const float biasu1 = bu1[n];
  const float biasu2 = bu2[n];
  __syncthreads();

  for (int v = 1; v < N_NODES; ++v) {
    const int ebase = (v - 1) * DEG;
    // ---- phase 1: gather z = [h[srcs] | attr | 0] -> LDS f16; h[v] -> sCat[0:128] ----
    for (int i = tid; i < 16 * KZ; i += 256) {
      int row = i / KZ, col = i % KZ;
      float val;
      if (col < HID) {
        int s = ei[ebase + row];                      // predecessor id
        val = h[s * 128 + col];
      } else if (col < HID + ED) {
        val = attr[(ebase + row) * ED + (col - HID)];
      } else {
        val = 0.0f;
      }
      sZ[i] = (_Float16)val;
    }
    if (tid < 128) sCat[tid] = (_Float16)h[v * 128 + tid];
    __syncthreads();

    // ---- phase 2: msg layer 1: [16,160]x[160,128], relu -> sMsg1 (f16) ----
    {
      v8f acc = {};
#pragma unroll
      for (int kt = 0; kt < 5; ++kt) {
        v16h a = load_a_frag(sZ, KZ, kt * 32, lane);
        v16h b = load_b_frag(sWm1T, KZ, w * 16, kt * 32, lane);
        acc = __builtin_amdgcn_wmma_f32_16x16x32_f16(false, a, false, b,
                                                     (short)0, acc, false, false);
      }
#pragma unroll
      for (int r = 0; r < 8; ++r) {
        float t = acc[r] + bias1;
        sMsg1[(rbase + r) * 128 + n] = (_Float16)(t > 0.0f ? t : 0.0f);
      }
    }
    __syncthreads();

    // ---- phase 3: msg layer 2: [16,128]x[128,128], relu, mean -> sCat[128:256] ----
    {
      v8f acc = {};
#pragma unroll
      for (int kt = 0; kt < 4; ++kt) {
        v16h a = load_a_frag(sMsg1, 128, kt * 32, lane);
        v16h b = load_b_frag(sWm2T, 128, w * 16, kt * 32, lane);
        acc = __builtin_amdgcn_wmma_f32_16x16x32_f16(false, a, false, b,
                                                     (short)0, acc, false, false);
      }
      float s = 0.0f;
#pragma unroll
      for (int r = 0; r < 8; ++r) {
        float t = acc[r] + bias2;
        s += (t > 0.0f ? t : 0.0f);
      }
      s += __shfl_xor(s, 16);                 // rows 0-7 + rows 8-15 (same column)
      if (lane < 16) sCat[128 + n] = (_Float16)(s * (1.0f / 16.0f)); // cnt==16, v>=1
    }
    __syncthreads();

    // ---- phase 4: update layer 1 as M=1 WMMA GEMV: cat[256] x Wu1 -> sU1 ----
    {
      v8f acc = {};
#pragma unroll
      for (int kt = 0; kt < 8; ++kt) {
        v16h a = load_a_frag_vec(sCat, kt * 32, lane);
        v16h b = load_b_frag(sWu1T, 256, w * 16, kt * 32, lane);
        acc = __builtin_amdgcn_wmma_f32_16x16x32_f16(false, a, false, b,
                                                     (short)0, acc, false, false);
      }
      if (lane < 16) {                        // row 0 of D = lanes 0..15, vgpr 0
        float t = acc[0] + biasu1;
        sU1[n] = (_Float16)(t > 0.0f ? t : 0.0f);
      }
    }
    __syncthreads();

    // ---- phase 5: update layer 2 as M=1 WMMA GEMV -> h[v] ----
    {
      v8f acc = {};
#pragma unroll
      for (int kt = 0; kt < 4; ++kt) {
        v16h a = load_a_frag_vec(sU1, kt * 32, lane);
        v16h b = load_b_frag(sWu2T, 128, w * 16, kt * 32, lane);
        acc = __builtin_amdgcn_wmma_f32_16x16x32_f16(false, a, false, b,
                                                     (short)0, acc, false, false);
      }
      if (lane < 16) {
        float t = acc[0] + biasu2;
        h[v * 128 + n] = (t > 0.0f ? t : 0.0f);
      }
    }
    __syncthreads();   // h[v] visible (workgroup scope) before later gathers
  }
}

extern "C" void kernel_launch(void* const* d_in, const int* in_sizes, int n_in,
                              void* d_out, int out_size, void* d_ws, size_t ws_size,
                              hipStream_t stream) {
  const float* x    = (const float*)d_in[0];
  const int*   ei   = (const int*)d_in[1];
  const float* attr = (const float*)d_in[2];
  const float* Wp   = (const float*)d_in[3];
  const float* bp   = (const float*)d_in[4];
  const float* Wm1  = (const float*)d_in[5];
  const float* bm1  = (const float*)d_in[6];
  const float* Wm2  = (const float*)d_in[7];
  const float* bm2  = (const float*)d_in[8];
  const float* Wu1  = (const float*)d_in[9];
  const float* bu1  = (const float*)d_in[10];
  const float* Wu2  = (const float*)d_in[11];
  const float* bu2  = (const float*)d_in[12];
  float* out = (float*)d_out;

  _Float16* wsH = (_Float16*)d_ws;

  prep_kernel<<<400, 256, 0, stream>>>(Wp, Wm1, Wm2, Wu1, Wu2, wsH);
  h0_kernel<<<N_NODES / 16, 256, 0, stream>>>(x, wsH /*WpT*/, bp, out);
  scan_kernel<<<1, 256, 0, stream>>>(out, ei, attr, bm1, bm2, bu1, bu2, wsH);
}
